// SparseMLP_60687887893042
// MI455X (gfx1250) — matrix operand order repaired and checked by reading, not compiled, not run on hardware
//
#include <hip/hip_runtime.h>

// ---------------------------------------------------------------------------
// CDNA5 WMMA (wave32): D(f32 16x16) = A(bf16 16x32) x B(bf16 32x16) + C
// ---------------------------------------------------------------------------
typedef __attribute__((ext_vector_type(16))) __bf16          v16bf;
typedef __attribute__((ext_vector_type(8)))  __bf16          v8bf;
typedef __attribute__((ext_vector_type(8)))  float           v8f;
typedef __attribute__((ext_vector_type(4)))  float           v4f;
typedef __attribute__((ext_vector_type(4)))  unsigned short  v4us;

__device__ __forceinline__ unsigned short f2bf_bits(float f) {
    union { float f; unsigned u; } v; v.f = f;
    unsigned r = v.u + 0x7FFFu + ((v.u >> 16) & 1u);   // round-to-nearest-even
    return (unsigned short)(r >> 16);
}

__device__ __forceinline__ v16bf cat8(v8bf lo, v8bf hi) {
    return __builtin_shufflevector(lo, hi, 0, 1, 2, 3, 4, 5, 6, 7,
                                            8, 9, 10, 11, 12, 13, 14, 15);
}

// 16-byte global -> LDS async copy (GLOBAL_LOAD_ASYNC_TO_LDS_B128, ASYNCcnt).
// %0 = LDS byte offset (low 32 bits of generic LDS pointer), %1 = 64-bit vaddr.
__device__ __forceinline__ void async_copy16(const void* gptr, void* lptr) {
    asm volatile("global_load_async_to_lds_b128 %0, %1, off"
                 :: "v"((unsigned)(uintptr_t)lptr), "v"(gptr)
                 : "memory");
}
__device__ __forceinline__ void wait_async0() {
    asm volatile("s_wait_asynccnt 0" ::: "memory");
}

// ---------------------------------------------------------------------------
// Weight prep: Wm = bf16(Ws * mask); W1p = bf16(W1) padded K 784->800;
// W2p = bf16(W2) padded N 10->16 (zero rows). All inputs read once -> NT loads.
// ---------------------------------------------------------------------------
__global__ __launch_bounds__(256) void k_prep(
    const float* __restrict__ W1, const float* __restrict__ Ws,
    const float* __restrict__ mask, const float* __restrict__ W2,
    unsigned short* __restrict__ W1p, unsigned short* __restrict__ Wm,
    unsigned short* __restrict__ W2p)
{
    const int total = 1024 * 1024;
    for (int i = blockIdx.x * 256 + threadIdx.x; i < total; i += gridDim.x * 256) {
        float ws = __builtin_nontemporal_load(&Ws[i]);
        float mk = __builtin_nontemporal_load(&mask[i]);
        Wm[i] = f2bf_bits(ws * mk);
        if (i < 1024 * 800) {
            int r = i / 800, c = i - r * 800;
            W1p[i] = (c < 784)
                ? f2bf_bits(__builtin_nontemporal_load(&W1[r * 784 + c]))
                : (unsigned short)0;
        }
        if (i < 16 * 1024) {
            int r = i >> 10, c = i & 1023;
            W2p[i] = (r < 10)
                ? f2bf_bits(__builtin_nontemporal_load(&W2[r * 1024 + c]))
                : (unsigned short)0;
        }
    }
}

// ---------------------------------------------------------------------------
// Layer 1: h1[M,1024](bf16) = relu(x[M,784](f32) @ W1p[1024,800]^T + b1)
// 256 thr = 8 waves; block tile 128M x 64N; wave tile 32x32; K step 32.
// x staged with NT float4 loads + branch-free tail (clamped addr + select);
// W staged via async global->LDS DMA. h1 stores default-RT (L2-resident for L2 reuse).
// ---------------------------------------------------------------------------
__global__ __launch_bounds__(256) void k_gemm1(
    const float* __restrict__ Af, const unsigned short* __restrict__ W,
    const float* __restrict__ bias, unsigned short* __restrict__ Out)
{
    constexpr int Ka = 784, Kw = 800, Nn = 1024;
    __shared__ unsigned short As[2][128][40];
    __shared__ unsigned short Bs[2][64][40];

    const int tid  = threadIdx.x;
    const int lane = tid & 31;
    const int wid  = tid >> 5;
    const int wm   = wid & 3;
    const int wn   = wid >> 2;
    const int lh   = lane & 15;
    const int kh   = lane >> 4;
    const int rowBase = blockIdx.y * 128;
    const int colBase = blockIdx.x * 64;

    auto stage = [&](int buf, int kk) {
        // x tile 128x32: four float4 chunks per thread, fp32 -> bf16.
        // Ka % 4 == 0, so each quad is fully in range or fully out; clamp the
        // address (always-legal NT load) and select zeros -> no EXEC branching.
        #pragma unroll
        for (int i = 0; i < 4; ++i) {
            int c = (i << 8) + tid;        // 0..1023
            int r = c >> 3, q = c & 7;     // row, quad-of-4-cols
            int k0 = kk + q * 4;
            const bool ok = (k0 < Ka);
            const int kc = ok ? k0 : 0;
            v4f f = __builtin_nontemporal_load(
                (const v4f*)&Af[(size_t)(rowBase + r) * Ka + kc]);
            v4us u = (v4us){ ok ? f2bf_bits(f.x) : (unsigned short)0,
                             ok ? f2bf_bits(f.y) : (unsigned short)0,
                             ok ? f2bf_bits(f.z) : (unsigned short)0,
                             ok ? f2bf_bits(f.w) : (unsigned short)0 };
            *(v4us*)&As[buf][r][q * 4] = u;
        }
        // W tile 64x32: one 16B async chunk per thread (Kw padded, in range).
        {
            int r = tid >> 2, oct = tid & 3;
            async_copy16(&W[(size_t)(colBase + r) * Kw + kk + oct * 8],
                         &Bs[buf][r][oct * 8]);
        }
    };

    v8f acc[2][2] = {};
    constexpr int nsteps = Kw / 32;        // 25

    stage(0, 0);
    wait_async0();
    __syncthreads();

    for (int s = 0; s < nsteps; ++s) {
        const int buf = s & 1;
        if (s + 2 < nsteps)
            __builtin_prefetch(&Af[(size_t)(rowBase + (tid >> 1)) * Ka + ((s + 2) << 5)], 0, 0);
        if (s + 1 < nsteps) stage(buf ^ 1, (s + 1) << 5);

        v16bf afrag[2], bfrag[2];
        #pragma unroll
        for (int im = 0; im < 2; ++im) {
            const int r = wm * 32 + im * 16 + lh;
            afrag[im] = cat8(*(const v8bf*)&As[buf][r][kh * 8],
                             *(const v8bf*)&As[buf][r][16 + kh * 8]);
        }
        #pragma unroll
        for (int jn = 0; jn < 2; ++jn) {
            const int r = wn * 32 + jn * 16 + lh;
            bfrag[jn] = cat8(*(const v8bf*)&Bs[buf][r][kh * 8],
                             *(const v8bf*)&Bs[buf][r][16 + kh * 8]);
        }
        #pragma unroll
        for (int im = 0; im < 2; ++im)
            #pragma unroll
            for (int jn = 0; jn < 2; ++jn)
                acc[im][jn] = __builtin_amdgcn_wmma_f32_16x16x32_bf16(
                    false, afrag[im], false, bfrag[jn],
                    (short)0, acc[im][jn], false, false);

        wait_async0();
        __syncthreads();
    }

    #pragma unroll
    for (int im = 0; im < 2; ++im) {
        const int rbase = rowBase + wm * 32 + im * 16 + kh * 8;
        #pragma unroll
        for (int jn = 0; jn < 2; ++jn) {
            const int col = colBase + wn * 32 + jn * 16 + lh;
            const float bv = bias[col];
            #pragma unroll
            for (int r = 0; r < 8; ++r) {
                float v = fmaxf(acc[im][jn][r] + bv, 0.0f);
                Out[(size_t)(rbase + r) * Nn + col] = f2bf_bits(v);
            }
        }
    }
}

// ---------------------------------------------------------------------------
// Layer 2: h2[M,1024](bf16) = relu(h1[M,1024](bf16) @ Wm[1024,1024]^T + bs)
// Both tiles staged entirely with async global->LDS DMA (no VGPR round-trip).
// h1 reads default-RT (served largely from the 192MB L2).
// ---------------------------------------------------------------------------
__global__ __launch_bounds__(256) void k_gemm2(
    const unsigned short* __restrict__ Ah, const unsigned short* __restrict__ W,
    const float* __restrict__ bias, unsigned short* __restrict__ Out)
{
    constexpr int K = 1024, Nn = 1024;
    __shared__ unsigned short As[2][128][40];
    __shared__ unsigned short Bs[2][64][40];

    const int tid  = threadIdx.x;
    const int lane = tid & 31;
    const int wid  = tid >> 5;
    const int wm   = wid & 3;
    const int wn   = wid >> 2;
    const int lh   = lane & 15;
    const int kh   = lane >> 4;
    const int rowBase = blockIdx.y * 128;
    const int colBase = blockIdx.x * 64;

    auto stage = [&](int buf, int kk) {
        #pragma unroll
        for (int i = 0; i < 2; ++i) {      // A tile: 512 chunks of 16B
            int c = (i << 8) + tid;
            int r = c >> 2, oct = c & 3;
            async_copy16(&Ah[(size_t)(rowBase + r) * K + kk + oct * 8],
                         &As[buf][r][oct * 8]);
        }
        {                                   // W tile: 256 chunks of 16B
            int r = tid >> 2, oct = tid & 3;
            async_copy16(&W[(size_t)(colBase + r) * K + kk + oct * 8],
                         &Bs[buf][r][oct * 8]);
        }
    };

    v8f acc[2][2] = {};
    constexpr int nsteps = K / 32;         // 32

    stage(0, 0);
    wait_async0();
    __syncthreads();

    for (int s = 0; s < nsteps; ++s) {
        const int buf = s & 1;
        if (s + 1 < nsteps) stage(buf ^ 1, (s + 1) << 5);

        v16bf afrag[2], bfrag[2];
        #pragma unroll
        for (int im = 0; im < 2; ++im) {
            const int r = wm * 32 + im * 16 + lh;
            afrag[im] = cat8(*(const v8bf*)&As[buf][r][kh * 8],
                             *(const v8bf*)&As[buf][r][16 + kh * 8]);
        }
        #pragma unroll
        for (int jn = 0; jn < 2; ++jn) {
            const int r = wn * 32 + jn * 16 + lh;
            bfrag[jn] = cat8(*(const v8bf*)&Bs[buf][r][kh * 8],
                             *(const v8bf*)&Bs[buf][r][16 + kh * 8]);
        }
        #pragma unroll
        for (int im = 0; im < 2; ++im)
            #pragma unroll
            for (int jn = 0; jn < 2; ++jn)
                acc[im][jn] = __builtin_amdgcn_wmma_f32_16x16x32_bf16(
                    false, afrag[im], false, bfrag[jn],
                    (short)0, acc[im][jn], false, false);

        wait_async0();
        __syncthreads();
    }

    #pragma unroll
    for (int im = 0; im < 2; ++im) {
        const int rbase = rowBase + wm * 32 + im * 16 + kh * 8;
        #pragma unroll
        for (int jn = 0; jn < 2; ++jn) {
            const int col = colBase + wn * 32 + jn * 16 + lh;
            const float bv = bias[col];
            #pragma unroll
            for (int r = 0; r < 8; ++r) {
                float v = fmaxf(acc[im][jn][r] + bv, 0.0f);
                Out[(size_t)(rbase + r) * Nn + col] = f2bf_bits(v);
            }
        }
    }
}

// ---------------------------------------------------------------------------
// Layer 3 + softmax: one 16x16 WMMA tile per wave vs padded W2 (cols>=10 zero),
// then in-register softmax over the 16-lane N dimension. NT output stores.
// ---------------------------------------------------------------------------
__global__ __launch_bounds__(256) void k_final(
    const unsigned short* __restrict__ H, const unsigned short* __restrict__ W2p,
    const float* __restrict__ b2, float* __restrict__ out)
{
    constexpr int K = 1024;
    __shared__ unsigned short As[2][128][40];
    __shared__ unsigned short Bs[2][16][40];

    const int tid  = threadIdx.x;
    const int lane = tid & 31;
    const int wid  = tid >> 5;
    const int lh   = lane & 15;
    const int kh   = lane >> 4;
    const int rowBase = blockIdx.x * 128;

    auto stage = [&](int buf, int kk) {
        #pragma unroll
        for (int i = 0; i < 2; ++i) {      // H tile: 512 chunks of 16B
            int c = (i << 8) + tid;
            int r = c >> 2, oct = c & 3;
            async_copy16(&H[(size_t)(rowBase + r) * K + kk + oct * 8],
                         &As[buf][r][oct * 8]);
        }
        if (tid < 64) {                     // W2 tile: 64 chunks of 16B
            int r = tid >> 2, oct = tid & 3;
            async_copy16(&W2p[(size_t)r * K + kk + oct * 8],
                         &Bs[buf][r][oct * 8]);
        }
    };

    v8f acc = {};
    constexpr int nsteps = K / 32;

    stage(0, 0);
    wait_async0();
    __syncthreads();

    for (int s = 0; s < nsteps; ++s) {
        const int buf = s & 1;
        if (s + 1 < nsteps) stage(buf ^ 1, (s + 1) << 5);

        const int ra = wid * 16 + lh;
        v16bf afrag = cat8(*(const v8bf*)&As[buf][ra][kh * 8],
                           *(const v8bf*)&As[buf][ra][16 + kh * 8]);
        v16bf bfrag = cat8(*(const v8bf*)&Bs[buf][lh][kh * 8],
                           *(const v8bf*)&Bs[buf][lh][16 + kh * 8]);

        acc = __builtin_amdgcn_wmma_f32_16x16x32_bf16(
            false, afrag, false, bfrag, (short)0, acc, false, false);

        wait_async0();
        __syncthreads();
    }

    const float bv = (lh < 10) ? b2[lh] : 0.0f;
    #pragma unroll
    for (int r = 0; r < 8; ++r) {
        float logit = acc[r] + bv;
        float x = (lh < 10) ? logit : -3.0e38f;
        float mx = x;
        #pragma unroll
        for (int m = 8; m >= 1; m >>= 1) mx = fmaxf(mx, __shfl_xor(mx, m, 16));
        float e = (lh < 10) ? __expf(x - mx) : 0.0f;
        float ssum = e;
        #pragma unroll
        for (int m = 8; m >= 1; m >>= 1) ssum += __shfl_xor(ssum, m, 16);
        if (lh < 10) {
            const int row = rowBase + wid * 16 + kh * 8 + r;
            __builtin_nontemporal_store(e / ssum, &out[(size_t)row * 10 + lh]);
        }
    }
}

// ---------------------------------------------------------------------------
// Launch: inputs = {x, W1, b1, Ws, bs, W2, b2, mask}; out = softmax [65536,10]
// ---------------------------------------------------------------------------
extern "C" void kernel_launch(void* const* d_in, const int* in_sizes, int n_in,
                              void* d_out, int out_size, void* d_ws, size_t ws_size,
                              hipStream_t stream) {
    (void)in_sizes; (void)n_in; (void)out_size; (void)ws_size;

    const float* x    = (const float*)d_in[0];
    const float* W1   = (const float*)d_in[1];
    const float* b1   = (const float*)d_in[2];
    const float* Ws   = (const float*)d_in[3];
    const float* bs   = (const float*)d_in[4];
    const float* W2   = (const float*)d_in[5];
    const float* b2   = (const float*)d_in[6];
    const float* mask = (const float*)d_in[7];
    float* out = (float*)d_out;

    char* ws = (char*)d_ws;
    unsigned short* W1p = (unsigned short*)(ws + 0);                      // 1024*800*2
    unsigned short* Wm  = (unsigned short*)(ws + 1638400);                // 1024*1024*2
    unsigned short* W2p = (unsigned short*)(ws + 3735552);                // 16*1024*2
    unsigned short* h1  = (unsigned short*)(ws + 3768320);                // 65536*1024*2
    unsigned short* h2  = (unsigned short*)(ws + 3768320 + 134217728ull); // 65536*1024*2

    k_prep<<<4096, 256, 0, stream>>>(W1, Ws, mask, W2, W1p, Wm, W2p);

    k_gemm1<<<dim3(1024 / 64, 65536 / 128), 256, 0, stream>>>(x, W1p, b1, h1);
    k_gemm2<<<dim3(1024 / 64, 65536 / 128), 256, 0, stream>>>(h1, Wm, bs, h2);
    k_final<<<65536 / 128, 256, 0, stream>>>(h2, W2p, b2, out);
}
